// YOLOX_43439299231991
// MI455X (gfx1250) — matrix operand order repaired and checked by reading, not compiled, not run on hardware
//
#include <hip/hip_runtime.h>
#include <hip/hip_bf16.h>
#include <cstdint>
#include <cstddef>

// ---------------------------------------------------------------------------
// CDNA5 (gfx1250) YOLOX head:
//   f16 WMMA implicit-GEMM convs with async global->LDS double-buffered
//   pipeline, fused BN+SiLU, + LDS-resident top-k/NMS.
// ---------------------------------------------------------------------------

typedef __attribute__((ext_vector_type(16))) _Float16 v16h;
typedef __attribute__((ext_vector_type(8)))  float    v8f;

#define BN_EPS   0.001f
#define NPRIOR   8500
#define KSEL     512
#define BOX_LIM  300
#define NCLS     80

__device__ __forceinline__ v8f wmma_f16(v16h a, v16h b, v8f c) {
  // D = A(16x32 f16) x B(32x16 f16) + C(16x16 f32)
  return __builtin_amdgcn_wmma_f32_16x16x32_f16(
      /*neg_a=*/false, a, /*neg_b=*/false, b,
      /*c_mod=*/(short)0, c, /*reuse_a=*/false, /*reuse_b=*/false);
}

// Async DMA: 16B per lane, global -> LDS, tracked by ASYNCcnt (no VGPR trip).
__device__ __forceinline__ void async_ld_b128(unsigned ldsOff, const void* g) {
  asm volatile("global_load_async_to_lds_b128 %0, %1, off"
               :: "v"(ldsOff), "v"((unsigned long long)(uintptr_t)g)
               : "memory");
}
__device__ __forceinline__ void wait_async0() {
  asm volatile("s_wait_asynccnt 0x0" ::: "memory");
}
__device__ __forceinline__ unsigned lds_off(const void* p) {
  return (unsigned)(uintptr_t)(__attribute__((address_space(3))) const void*)p;
}

// ---------------------------------------------------------------------------
// Fold BN into conv weights; repack as Wp[tap][cout][cin] (K contiguous) f16.
// ---------------------------------------------------------------------------
__global__ void prep_cba(const float* __restrict__ w, const float* __restrict__ gamma,
                         const float* __restrict__ beta, const float* __restrict__ mean,
                         const float* __restrict__ var,
                         _Float16* __restrict__ wOut, float* __restrict__ biasOut,
                         int cout, int cin, int kk) {
  int i = blockIdx.x * blockDim.x + threadIdx.x;
  int total = cout * cin * kk;
  if (i >= total) return;
  int t = i % kk;
  int rest = i / kk;
  int ci = rest % cin;
  int co = rest / cin;
  float scale = gamma[co] * rsqrtf(var[co] + BN_EPS);
  wOut[((size_t)t * cout + co) * cin + ci] =
      (_Float16)(w[((size_t)co * cin + ci) * kk + t] * scale);
  if (ci == 0 && t == 0) biasOut[co] = beta[co] - mean[co] * scale;
}

// ---------------------------------------------------------------------------
// Implicit-GEMM conv (1x1 or 3x3 SAME), fused bias + SiLU.
// Block: 256 thr = 8 waves. Tile: M=128 spatial x N=128 cout, double-buffered
// LDS, async global->LDS staging overlapped with WMMA.
// Wave (wm, wn) owns 32x64 -> 2x4 WMMA 16x16 f32 accumulators.
// ---------------------------------------------------------------------------
__global__ void __launch_bounds__(256)
conv_gemm(const float* __restrict__ srcF32,     // NCHW fp32 (stem) or null
          const _Float16* __restrict__ srcF16,  // NHWC f16 or null
          const _Float16* __restrict__ Wp,      // [taps][Cout][Cin] f16
          const float* __restrict__ bias,       // [Cout]
          _Float16* __restrict__ dst,           // NHWC f16 [M][Cout]
          int Bn, int H, int W, int Cin, int Cout, int taps) {
  const int Mtot = Bn * H * W;
  const int M0 = blockIdx.x * 128;
  const int N0 = blockIdx.y * 128;

  // 40-half rows (80B): keeps b128 fragment gathers 16B-aligned, kills
  // bank conflicts. Double buffered for the async pipeline.
  __shared__ _Float16 ldsA[2][128 * 40];
  __shared__ _Float16 ldsB[2][128 * 40];

  const int tid  = threadIdx.x;
  const int wave = tid >> 5;
  const int lane = tid & 31;
  const int wm = wave & 3;       // M sub-tile (x32 rows)
  const int wn = wave >> 2;      // N sub-tile (x64 cols)
  const int halfL = lane >> 4;   // lane group (K-half select)
  const int ml = lane & 15;

  const int kSteps = Cin >> 5;
  const int iters = taps * kSteps;

  v8f acc[2][4] = {};

  // ---- stage tile `it` into LDS buffer `buf` (async where possible) ----
  auto stage = [&](int it, int buf) {
    const int tap = (taps == 9) ? (it / kSteps) : 0;
    const int k0 = (it - tap * kSteps) << 5;
    const int dy = (taps == 9) ? (tap / 3 - 1) : 0;
    const int dx = (taps == 9) ? (tap % 3 - 1) : 0;
    // A: 128 rows x 32 K-halves; thread -> (row, 16-half segment)
    {
      const int rr = tid >> 1, seg = tid & 1;
      _Float16* out = &ldsA[buf][rr * 40 + seg * 16];
      const int m = M0 + rr;
      bool ok = (m < Mtot);
      int b = 0, ys = 0, xs = 0;
      if (ok) {
        const int hw = H * W;
        b = m / hw;
        const int r2 = m - b * hw;
        const int y = r2 / W, x = r2 - y * W;
        ys = y + dy; xs = x + dx;
        ok = (ys >= 0) && (ys < H) && (xs >= 0) && (xs < W);
      }
      if (ok) {
        if (srcF32) {  // stem: NCHW fp32 channel-strided gather + cvt (sync)
          const size_t hw = (size_t)H * W;
          const float* p = srcF32 + ((size_t)b * Cin + (k0 + seg * 16)) * hw +
                           (size_t)ys * W + xs;
          __builtin_prefetch(p + 16 * hw, 0, 1);
#pragma unroll
          for (int c = 0; c < 16; ++c) out[c] = (_Float16)p[c * hw];
        } else {       // NHWC f16: 32B contiguous -> 2 async b128 DMAs
          const _Float16* p =
              srcF16 + ((size_t)((b * H + ys) * W + xs)) * Cin + k0 + seg * 16;
          const unsigned lo = lds_off(out);
          async_ld_b128(lo, p);
          async_ld_b128(lo + 16u, p + 8);
        }
      } else {
#pragma unroll
        for (int c = 0; c < 16; ++c) out[c] = (_Float16)0.0f;
      }
    }
    // B: 128 cout rows x 32 K-halves (K contiguous) -> 2 async b128 DMAs
    {
      const int n = tid >> 1, seg = tid & 1;
      const _Float16* p =
          Wp + ((size_t)tap * Cout + (N0 + n)) * Cin + k0 + seg * 16;
      _Float16* out = &ldsB[buf][n * 40 + seg * 16];
      const unsigned lo = lds_off(out);
      async_ld_b128(lo, p);
      async_ld_b128(lo + 16u, p + 8);
    }
  };

  // ---- consume LDS buffer: fragment gathers + 8 chained WMMAs ----
  auto compute = [&](int buf) {
    v16h a0, a1;
    {
      // A 16x32: lane-half 0 -> K 0..7 & 16..23 ; half 1 -> K 8..15 & 24..31
      const _Float16* pa = &ldsA[buf][(wm * 32 + ml) * 40 + halfL * 8];
      ((float4*)&a0)[0] = *(const float4*)pa;
      ((float4*)&a0)[1] = *(const float4*)(pa + 16);
      pa += 16 * 40;
      ((float4*)&a1)[0] = *(const float4*)pa;
      ((float4*)&a1)[1] = *(const float4*)(pa + 16);
    }
#pragma unroll
    for (int ni = 0; ni < 4; ++ni) {
      // B 32x16: lanes 0-15 -> K 0..7 & 16..23 ; lanes 16-31 -> K 8..15 & 24..31
      const _Float16* pb = &ldsB[buf][(wn * 64 + ni * 16 + ml) * 40 + halfL * 8];
      v16h bf;
      ((float4*)&bf)[0] = *(const float4*)pb;
      ((float4*)&bf)[1] = *(const float4*)(pb + 16);
      acc[0][ni] = wmma_f16(a0, bf, acc[0][ni]);
      acc[1][ni] = wmma_f16(a1, bf, acc[1][ni]);
    }
  };

  // ---- software pipeline: stage(it+1) overlaps compute(it) ----
  stage(0, 0);
  for (int it = 0; it < iters; ++it) {
    const int buf = it & 1;
    wait_async0();       // our async writes to LDS have landed
    __syncthreads();     // everyone's have; previous reads also done
    if (it + 1 < iters) stage(it + 1, buf ^ 1);
    compute(buf);
  }

  // ---- epilogue: bias + SiLU, store NHWC f16 ----
#pragma unroll
  for (int mi = 0; mi < 2; ++mi) {
#pragma unroll
    for (int ni = 0; ni < 4; ++ni) {
      const int mBase = M0 + wm * 32 + mi * 16;
      const int nBase = N0 + wn * 64 + ni * 16;
#pragma unroll
      for (int r = 0; r < 8; ++r) {
        const int m = mBase + halfL * 8 + r;  // C/D: lanes16-31 -> M+8
        const int n = nBase + ml;
        if (m < Mtot) {
          float v = acc[mi][ni][r] + bias[n];
          v = v / (1.0f + __expf(-v));  // SiLU
          dst[(size_t)m * Cout + n] = (_Float16)v;
        }
      }
    }
  }
}

// ---------------------------------------------------------------------------
// Output heads (1x1, 85ch) + YOLO decode into yolo[(b, anchor, 88)].
// 88 = xy(2) wh(2) obj(1) cls(80) stride(1) anchor(2)
// ---------------------------------------------------------------------------
__global__ void head_decode(const _Float16* __restrict__ reg,
                            const _Float16* __restrict__ cls,
                            const float* __restrict__ regW, const float* __restrict__ regB,
                            const float* __restrict__ objW, const float* __restrict__ objB,
                            const float* __restrict__ clsW, const float* __restrict__ clsB,
                            float* __restrict__ yolo, int Bn, int H, int W,
                            float s, int anchorBase, int anchorsTot) {
  const int i = blockIdx.x * blockDim.x + threadIdx.x;
  const int total = Bn * H * W * 88;
  if (i >= total) return;
  const int oc = i % 88;
  const int rest = i / 88;
  const int hw = H * W;
  const int b = rest / hw;
  const int pos = rest - b * hw;
  const int row = pos / W, col = pos - row * W;

  float v = 0.0f;
  if (oc < 85) {
    const size_t abase = ((size_t)b * hw + pos) * 256;
    const _Float16* a = (oc < 5) ? (reg + abase) : (cls + abase);
    const float* wrow;
    float b0;
    if (oc < 4)      { wrow = regW + oc * 256;       b0 = regB[oc]; }
    else if (oc == 4){ wrow = objW;                  b0 = objB[0]; }
    else             { wrow = clsW + (oc - 5) * 256; b0 = clsB[oc - 5]; }
    float acc = 0.0f;
#pragma unroll 8
    for (int c = 0; c < 256; ++c) acc += (float)a[c] * wrow[c];
    v = acc + b0;
  }
  float out;
  if (oc == 0)            out = (v + 0.5f + (float)col) * s;
  else if (oc == 1)       out = (v + 0.5f + (float)row) * s;
  else if (oc == 2 || oc == 3) out = v * v * s;
  else if (oc < 85)       out = v;            // obj / cls logits, raw
  else if (oc == 85)      out = s;            // stride col
  else if (oc == 86)      out = ((float)col + 0.5f) * s;
  else                    out = ((float)row + 0.5f) * s;
  yolo[((size_t)b * anchorsTot + anchorBase + pos) * 88 + oc] = out;
}

// ---------------------------------------------------------------------------
// Per-image top-512 + class-offset NMS + ordering. One block per image.
// Everything lives in LDS (~97KB of the WGP's 320KB), dynamic allocation.
// ---------------------------------------------------------------------------
__global__ void __launch_bounds__(512)
nms_kernel(const float* __restrict__ yolo, float* __restrict__ boxesOut,
           float* __restrict__ scoresOut, float* __restrict__ labelsOut,
           float* __restrict__ validOut) {
  extern __shared__ unsigned char smemRaw[];
  float* sScore    = (float*)smemRaw;          // [8512]
  float* sSelScore = sScore + 8512;            // [512]
  float* sBox      = sSelScore + 512;          // [512*4]
  float* sOb       = sBox + 2048;              // [512*4]
  float* sRedS     = sOb + 2048;               // [512]
  float* sArea     = sRedS + 512;              // [512]
  int* sSelIdx = (int*)(sArea + 512);          // [512]
  int* sLabel  = sSelIdx + 512;                // [512]
  int* sKeep   = sLabel + 512;                 // [512]
  int* sScan   = sKeep + 512;                  // [512]
  int* sRedI   = sScan + 512;                  // [512]
  unsigned* sMask = (unsigned*)(sRedI + 512);  // [512*16]

  const int b = blockIdx.x, tid = threadIdx.x;
  const float* base = yolo + (size_t)b * NPRIOR * 88;

  // masked sigmoid(obj) scores
  for (int j = tid; j < NPRIOR; j += 512) {
    float l = base[(size_t)j * 88 + 4];
    float s = 1.0f / (1.0f + __expf(-l));
    sScore[j] = (s > 0.5f) ? s : -1.0f;
  }
  __syncthreads();

  // top-512 by iterative argmax (stable: ties -> lowest index)
  for (int it = 0; it < KSEL; ++it) {
    float bs = -4.0f;
    int bi = 0;
    for (int j = tid; j < NPRIOR; j += 512) {
      float s = sScore[j];
      if (s > bs) { bs = s; bi = j; }
    }
    sRedS[tid] = bs; sRedI[tid] = bi;
    __syncthreads();
    for (int off = 256; off > 0; off >>= 1) {
      if (tid < off) {
        float s2 = sRedS[tid + off]; int i2 = sRedI[tid + off];
        if (s2 > sRedS[tid] || (s2 == sRedS[tid] && i2 < sRedI[tid])) {
          sRedS[tid] = s2; sRedI[tid] = i2;
        }
      }
      __syncthreads();
    }
    if (tid == 0) {
      sSelIdx[it] = sRedI[0];
      sSelScore[it] = sRedS[0];
      sScore[sRedI[0]] = -4.0f;  // mark taken
    }
    __syncthreads();
  }

  // candidates: boxes, labels, validity, max-coord contribution
  {
    const int idx = sSelIdx[tid];
    const float* r = base + (size_t)idx * 88;
    float cx = r[0], cy = r[1], w = r[2], h = r[3];
    float x0 = cx - w * 0.5f, y0 = cy - h * 0.5f;
    float x1 = cx + w * 0.5f, y1 = cy + h * 0.5f;
    sBox[tid * 4 + 0] = x0; sBox[tid * 4 + 1] = y0;
    sBox[tid * 4 + 2] = x1; sBox[tid * 4 + 3] = y1;
    float bl = r[5]; int lab = 0;
    for (int c = 1; c < NCLS; ++c) {
      float vv = r[5 + c];
      if (vv > bl) { bl = vv; lab = c; }
    }
    sLabel[tid] = lab;
    const int valid = (sSelScore[tid] > 0.5f) ? 1 : 0;
    sKeep[tid] = valid;
    float m4 = fmaxf(fmaxf(x0, y0), fmaxf(x1, y1));
    sRedS[tid] = valid ? m4 : 0.0f;
  }
  __syncthreads();
  for (int off = 256; off > 0; off >>= 1) {
    if (tid < off) sRedS[tid] = fmaxf(sRedS[tid], sRedS[tid + off]);
    __syncthreads();
  }
  const float maxc = sRedS[0];
  {
    const float o = (float)sLabel[tid] * (maxc + 1.0f);
    float x0 = sBox[tid * 4 + 0] + o, y0 = sBox[tid * 4 + 1] + o;
    float x1 = sBox[tid * 4 + 2] + o, y1 = sBox[tid * 4 + 3] + o;
    sOb[tid * 4 + 0] = x0; sOb[tid * 4 + 1] = y0;
    sOb[tid * 4 + 2] = x1; sOb[tid * 4 + 3] = y1;
    sArea[tid] = (x1 - x0) * (y1 - y0);
  }
  __syncthreads();

  // IoU > 0.5 bitmask: thread j vs all i
  {
    const float jx0 = sOb[tid * 4 + 0], jy0 = sOb[tid * 4 + 1];
    const float jx1 = sOb[tid * 4 + 2], jy1 = sOb[tid * 4 + 3];
    const float ja = sArea[tid];
    for (int w = 0; w < 16; ++w) {
      unsigned bits = 0u;
      for (int q = 0; q < 32; ++q) {
        const int i = w * 32 + q;
        float lx = fmaxf(sOb[i * 4 + 0], jx0), ly = fmaxf(sOb[i * 4 + 1], jy0);
        float rx = fminf(sOb[i * 4 + 2], jx1), ry = fminf(sOb[i * 4 + 3], jy1);
        float iw = fmaxf(rx - lx, 0.0f), ih = fmaxf(ry - ly, 0.0f);
        float inter = iw * ih;
        float iou = inter / (sArea[i] + ja - inter + 1e-9f);
        if (iou > 0.5f) bits |= (1u << q);
      }
      sMask[tid * 16 + w] = bits;
    }
  }
  __syncthreads();

  // sequential suppression (matches fori_loop semantics exactly)
  for (int i = 0; i < KSEL; ++i) {
    const int ki = sKeep[i];
    if (ki && tid > i) {
      if ((sMask[tid * 16 + (i >> 5)] >> (i & 31)) & 1u) sKeep[tid] = 0;
    }
    __syncthreads();
  }

  // ordering: kept (by index) first, then unkept; take first 300
  sScan[tid] = sKeep[tid];
  __syncthreads();
  for (int off = 1; off < KSEL; off <<= 1) {
    int v = (tid >= off) ? sScan[tid - off] : 0;
    __syncthreads();
    sScan[tid] += v;
    __syncthreads();
  }
  {
    const int inc = sScan[tid];
    const int exc = inc - sKeep[tid];
    const int totalKept = sScan[KSEL - 1];
    const int pos = sKeep[tid] ? exc : totalKept + (tid - exc);
    if (pos < BOX_LIM) {
      boxesOut[((size_t)b * BOX_LIM + pos) * 4 + 0] = sBox[tid * 4 + 0];
      boxesOut[((size_t)b * BOX_LIM + pos) * 4 + 1] = sBox[tid * 4 + 1];
      boxesOut[((size_t)b * BOX_LIM + pos) * 4 + 2] = sBox[tid * 4 + 2];
      boxesOut[((size_t)b * BOX_LIM + pos) * 4 + 3] = sBox[tid * 4 + 3];
      scoresOut[b * BOX_LIM + pos] = sSelScore[tid];
      labelsOut[b * BOX_LIM + pos] = (float)sLabel[tid];
      validOut[b * BOX_LIM + pos]  = sKeep[tid] ? 1.0f : 0.0f;
    }
  }
}

// ---------------------------------------------------------------------------
// Host orchestration
// ---------------------------------------------------------------------------
extern "C" void kernel_launch(void* const* d_in, const int* in_sizes, int n_in,
                              void* d_out, int out_size, void* d_ws, size_t ws_size,
                              hipStream_t stream) {
  (void)in_sizes; (void)n_in; (void)out_size; (void)ws_size;
  const int Bn = 8;
  const int IN_CH[4] = {256, 512, 768, 1024};
  const int HS[4]    = {80, 40, 20, 10};
  const int STR[4]   = {8, 16, 32, 64};
  const int ABASE[4] = {0, 6400, 8000, 8400};
  const int ATOT = NPRIOR;

  const float* const* in = (const float* const*)d_in;

  float* yolo      = (float*)d_out;
  float* boxesOut  = yolo + (size_t)Bn * ATOT * 88;
  float* scoresOut = boxesOut + (size_t)Bn * BOX_LIM * 4;
  float* labelsOut = scoresOut + (size_t)Bn * BOX_LIM;
  float* validOut  = labelsOut + (size_t)Bn * BOX_LIM;

  // workspace carve (256B aligned)
  char* ws = (char*)d_ws;
  auto wsAlloc = [&](size_t bytes) -> char* {
    char* p = ws;
    ws += (bytes + 255) & ~(size_t)255;
    return p;
  };
  _Float16* stemW[4]; float* stemB[4];
  _Float16* convW[4][4]; float* convB[4][4];
  for (int L = 0; L < 4; ++L) {
    stemW[L] = (_Float16*)wsAlloc((size_t)IN_CH[L] * 256 * sizeof(_Float16));
    stemB[L] = (float*)wsAlloc(256 * sizeof(float));
    for (int j = 0; j < 4; ++j) {
      convW[L][j] = (_Float16*)wsAlloc((size_t)9 * 256 * 256 * sizeof(_Float16));
      convB[L][j] = (float*)wsAlloc(256 * sizeof(float));
    }
  }
  const size_t maxM = (size_t)Bn * 80 * 80;  // largest level
  _Float16* actStem = (_Float16*)wsAlloc(maxM * 256 * sizeof(_Float16));
  _Float16* actTmp  = (_Float16*)wsAlloc(maxM * 256 * sizeof(_Float16));
  _Float16* actReg  = (_Float16*)wsAlloc(maxM * 256 * sizeof(_Float16));
  _Float16* actCls  = (_Float16*)wsAlloc(maxM * 256 * sizeof(_Float16));

  // 1) fold BN + repack weights
  for (int L = 0; L < 4; ++L) {
    const int pb = 4 + L * 31;
    {
      int total = 256 * IN_CH[L] * 1;
      prep_cba<<<(total + 255) / 256, 256, 0, stream>>>(
          in[pb + 0], in[pb + 1], in[pb + 2], in[pb + 3], in[pb + 4],
          stemW[L], stemB[L], 256, IN_CH[L], 1);
    }
    for (int j = 0; j < 4; ++j) {
      const int o = pb + 5 + 5 * j;
      int total = 256 * 256 * 9;
      prep_cba<<<(total + 255) / 256, 256, 0, stream>>>(
          in[o + 0], in[o + 1], in[o + 2], in[o + 3], in[o + 4],
          convW[L][j], convB[L][j], 256, 256, 9);
    }
  }

  // 2) per-level conv pipeline + head decode
  for (int L = 0; L < 4; ++L) {
    const int pb = 4 + L * 31;
    const int H = HS[L], W = HS[L];
    const int M = Bn * H * W;
    dim3 g((M + 127) / 128, 256 / 128);

    // stem 1x1 (fp32 NCHW in)
    conv_gemm<<<g, 256, 0, stream>>>(in[L], nullptr, stemW[L], stemB[L],
                                     actStem, Bn, H, W, IN_CH[L], 256, 1);
    // reg branch
    conv_gemm<<<g, 256, 0, stream>>>(nullptr, actStem, convW[L][0], convB[L][0],
                                     actTmp, Bn, H, W, 256, 256, 9);
    conv_gemm<<<g, 256, 0, stream>>>(nullptr, actTmp, convW[L][1], convB[L][1],
                                     actReg, Bn, H, W, 256, 256, 9);
    // cls branch
    conv_gemm<<<g, 256, 0, stream>>>(nullptr, actStem, convW[L][2], convB[L][2],
                                     actTmp, Bn, H, W, 256, 256, 9);
    conv_gemm<<<g, 256, 0, stream>>>(nullptr, actTmp, convW[L][3], convB[L][3],
                                     actCls, Bn, H, W, 256, 256, 9);
    // heads + decode into yolo block of d_out
    {
      int total = M * 88;
      head_decode<<<(total + 255) / 256, 256, 0, stream>>>(
          actReg, actCls,
          in[pb + 25], in[pb + 26],   // reg_out w,b
          in[pb + 27], in[pb + 28],   // obj_out w,b
          in[pb + 29], in[pb + 30],   // cls_out w,b
          yolo, Bn, H, W, (float)STR[L], ABASE[L], ATOT);
    }
  }

  // 3) per-image top-k + NMS + ordering
  {
    const size_t smem =
        sizeof(float) * (8512 + 512 + 2048 + 2048 + 512 + 512) +  // floats
        sizeof(int) * (512 * 5) +                                 // ints
        sizeof(unsigned) * (512 * 16);                            // iou mask
    nms_kernel<<<Bn, 512, smem, stream>>>(yolo, boxesOut, scoresOut,
                                          labelsOut, validOut);
  }
}
